// NeuroMUSX_27401891349153
// MI455X (gfx1250) — matrix operand order repaired and checked by reading, not compiled, not run on hardware
//
#include <hip/hip_runtime.h>
#include <hip/hip_bf16.h>

// ---------------------------------------------------------------------------
// GATv2 stack for MI455X (gfx1250, wave32).
//  - Node transforms on the WMMA pipe (v_wmma_f32_16x16x32_f16, f32 accum).
//    Activations are stored once in f16 (they are only consumed as WMMA A
//    operands); weights are pre-transposed to n-major f16 so every WMMA
//    fragment is two contiguous 16B loads. All tile buffers are padded to a
//    multiple of 64 rows so GEMM loads/stores need no per-lane guards
//    (EXEC stays all-1s through the whole kernel, as WMMA requires).
//  - Edge softmax phases are gather/atomic bound and stay f32
//    (ordered-uint atomicMax for segment max, f32 atomicAdd for sums).
// ---------------------------------------------------------------------------

typedef __attribute__((ext_vector_type(16))) _Float16 v16h;
typedef __attribute__((ext_vector_type(8)))  _Float16 v8h;
typedef __attribute__((ext_vector_type(8)))  float    v8f;

#define NNODES 25000
#define NEDGES 400000
#define NGRAPH 64
#define NITER  10
#define NPAD   25024   // NNODES rounded up to multiple of 64

__device__ __forceinline__ v16h cat16(v8h lo, v8h hi) {
  return __builtin_shufflevector(lo, hi, 0, 1, 2, 3, 4, 5, 6, 7,
                                 8, 9, 10, 11, 12, 13, 14, 15);
}

// ---- weight prep: f32 [K,Nout] row-major -> f16 [Nout,K] (n-major) ---------
__global__ void k_prep_w64(const float* __restrict__ Wl, const float* __restrict__ Wr,
                           _Float16* __restrict__ wlt, _Float16* __restrict__ wrt) {
  int tid = blockIdx.x * blockDim.x + threadIdx.x;   // 4096 threads
  int n = tid >> 6, k = tid & 63;
  wlt[tid] = (_Float16)Wl[k * 64 + n];
  wrt[tid] = (_Float16)Wr[k * 64 + n];
}

__global__ void k_prep_w16(const float* __restrict__ Wl, const float* __restrict__ Wr,
                           _Float16* __restrict__ wlt, _Float16* __restrict__ wrt) {
  int tid = blockIdx.x * blockDim.x + threadIdx.x;   // 1024 threads
  int n = tid >> 6, k = tid & 63;
  wlt[tid] = (_Float16)Wl[k * 16 + n];
  wrt[tid] = (_Float16)Wr[k * 16 + n];
}

// ---- node GEMM, 64->64: one wave = 16x16 tile, K=64 = 2 WMMA ---------------
// Fragment layouts per CDNA5 ISA 7.12.2 (wave32):
//   A 16x32 f16: lane<16 -> K {0..7,16..23}, lane>=16 -> K {8..15,24..31}
//   B 32x16 f16: same K order as A (lane<16 -> K {0..7,16..23}, lane>=16 -> K {8..15,24..31})
//   C/D f32:     element r -> M = r + 8*(lane>=16), N = lane&15
__global__ __launch_bounds__(256) void k_gemm_node64(
    const _Float16* __restrict__ X16,                 // [NPAD,64]
    const _Float16* __restrict__ Wlt, const float* __restrict__ bl,
    const _Float16* __restrict__ Wrt, const float* __restrict__ br,
    float* __restrict__ XL, float* __restrict__ XR) { // [NPAD,64]
  const int wave = threadIdx.x >> 5;
  const int lane = threadIdx.x & 31;
  const int hl   = lane >> 4;
  const int l15  = lane & 15;
  const int m0   = blockIdx.x * 16;
  const _Float16* Wt  = (wave < 4) ? Wlt : Wrt;
  const float*    bia = (wave < 4) ? bl : br;
  float*          OUT = (wave < 4) ? XL : XR;
  const int n = (wave & 3) * 16 + l15;
  const _Float16* arow = X16 + (size_t)(m0 + l15) * 64 + hl * 8;
  const _Float16* bcol = Wt + (size_t)n * 64 + hl * 8;
  v8f acc = {};
#pragma unroll
  for (int s = 0; s < 2; ++s) {
    v8h alo = *(const v8h*)(arow + s * 32);
    v8h ahi = *(const v8h*)(arow + s * 32 + 16);
    v8h blo = *(const v8h*)(bcol + s * 32);
    v8h bhi = *(const v8h*)(bcol + s * 32 + 16);
    acc = __builtin_amdgcn_wmma_f32_16x16x32_f16(false, cat16(alo, ahi),
                                                 false, cat16(blo, bhi),
                                                 (short)0, acc, false, false);
  }
  const float bv = bia[n];
  float* outp = OUT + (size_t)(m0 + hl * 8) * 64 + n;
#pragma unroll
  for (int r = 0; r < 8; ++r) outp[(size_t)r * 64] = acc[r] + bv;
}

// ---- node GEMM, 64->16 (final / sbs heads) ---------------------------------
__global__ __launch_bounds__(256) void k_gemm_node16(
    const _Float16* __restrict__ X16,                 // [NPAD,64]
    const _Float16* __restrict__ Wlt, const float* __restrict__ bl,
    const _Float16* __restrict__ Wrt, const float* __restrict__ br,
    float* __restrict__ XL, float* __restrict__ XR) { // [NPAD,16]
  const int wave = threadIdx.x >> 5;
  const int lane = threadIdx.x & 31;
  const int hl   = lane >> 4;
  const int l15  = lane & 15;
  const int m0   = blockIdx.x * 64 + (wave & 3) * 16;
  const _Float16* Wt  = (wave < 4) ? Wlt : Wrt;
  const float*    bia = (wave < 4) ? bl : br;
  float*          OUT = (wave < 4) ? XL : XR;
  const _Float16* arow = X16 + (size_t)(m0 + l15) * 64 + hl * 8;
  const _Float16* bcol = Wt + (size_t)l15 * 64 + hl * 8;
  v8f acc = {};
#pragma unroll
  for (int s = 0; s < 2; ++s) {
    v8h alo = *(const v8h*)(arow + s * 32);
    v8h ahi = *(const v8h*)(arow + s * 32 + 16);
    v8h blo = *(const v8h*)(bcol + s * 32);
    v8h bhi = *(const v8h*)(bcol + s * 32 + 16);
    acc = __builtin_amdgcn_wmma_f32_16x16x32_f16(false, cat16(alo, ahi),
                                                 false, cat16(blo, bhi),
                                                 (short)0, acc, false, false);
  }
  const float bv = bia[l15];
  float* outp = OUT + (size_t)(m0 + hl * 8) * 16 + l15;
#pragma unroll
  for (int r = 0; r < 8; ++r) outp[(size_t)r * 16] = acc[r] + bv;
}

// ---- self-loop edge-attr mean ----------------------------------------------
__global__ void k_selfloop_accum(const int* __restrict__ ei,
                                 const float* __restrict__ eattr,
                                 float* __restrict__ cnt, float* __restrict__ esum,
                                 int E) {
  int e = blockIdx.x * blockDim.x + threadIdx.x;
  if (e >= E) return;
  int d = ei[E + e];
  atomicAdd(&cnt[d], 1.0f);
  atomicAdd(&esum[2 * d + 0], eattr[2 * e + 0]);
  atomicAdd(&esum[2 * d + 1], eattr[2 * e + 1]);
}

__global__ void k_selfloop_finish(const float* __restrict__ cnt,
                                  const float* __restrict__ esum,
                                  float* __restrict__ loop_attr, int N) {
  int n = blockIdx.x * blockDim.x + threadIdx.x;
  if (n >= N) return;
  float c = fmaxf(cnt[n], 1.0f);
  loop_attr[2 * n + 0] = esum[2 * n + 0] / c;
  loop_attr[2 * n + 1] = esum[2 * n + 1] / c;
}

// ---- layer-0 node transform (K=2, not worth WMMA) --------------------------
__global__ void k_transform2(const float* __restrict__ x,
                             const float* __restrict__ Wl, const float* __restrict__ bl,
                             const float* __restrict__ Wr, const float* __restrict__ br,
                             float* __restrict__ xl, float* __restrict__ xr, int N) {
  int tid = blockIdx.x * blockDim.x + threadIdx.x;
  if (tid >= N * 64) return;
  int n = tid >> 6, f = tid & 63;
  float x0 = x[2 * n], x1 = x[2 * n + 1];
  xl[(size_t)n * 64 + f] = x0 * Wl[f] + x1 * Wl[64 + f] + bl[f];
  xr[(size_t)n * 64 + f] = x0 * Wr[f] + x1 * Wr[64 + f] + br[f];
}

// ---- order-preserving float<->uint for atomicMax ---------------------------
__device__ __forceinline__ unsigned f2ord(float f) {
  unsigned u = __float_as_uint(f);
  return (u & 0x80000000u) ? ~u : (u | 0x80000000u);
}
__device__ __forceinline__ float ord2f(unsigned u) {
  unsigned b = (u & 0x80000000u) ? (u ^ 0x80000000u) : ~u;
  return __uint_as_float(b);
}

// ---- edge pass 1: logits + segment max (thread per edge x head) ------------
template <int C>
__global__ void k_edge_logit(const int* __restrict__ ei,
                             const float* __restrict__ eattr,
                             const float* __restrict__ loop_attr,
                             const float* __restrict__ xl, const float* __restrict__ xr,
                             const float* __restrict__ We, const float* __restrict__ att,
                             float* __restrict__ logit, unsigned* __restrict__ lmax,
                             int E, int Etot) {
  long tid = (long)blockIdx.x * blockDim.x + threadIdx.x;
  int e = (int)(tid >> 4), hh = (int)(tid & 15);
  if (e >= Etot) return;
  int s, d; float a0, a1;
  if (e < E) { s = ei[e]; d = ei[E + e]; a0 = eattr[2 * e]; a1 = eattr[2 * e + 1]; }
  else       { s = d = e - E;           a0 = loop_attr[2 * s]; a1 = loop_attr[2 * s + 1]; }
  const int HC = 16 * C;
  float lg = 0.0f;
#pragma unroll
  for (int c = 0; c < C; ++c) {
    int f = hh * C + c;
    float v = xl[(size_t)s * HC + f] + xr[(size_t)d * HC + f]
            + a0 * We[f] + a1 * We[HC + f];
    v = (v > 0.0f) ? v : 0.2f * v;              // LeakyReLU(0.2)
    lg += v * att[f];
  }
  logit[(size_t)e * 16 + hh] = lg;
  atomicMax(&lmax[(size_t)d * 16 + hh], f2ord(lg));
}

// ---- edge pass 2: exp + segment sums ---------------------------------------
template <int C>
__global__ void k_edge_soft(const int* __restrict__ ei,
                            const float* __restrict__ logit,
                            const unsigned* __restrict__ lmax,
                            const float* __restrict__ xl,
                            float* __restrict__ den, float* __restrict__ out_num,
                            int E, int Etot) {
  long tid = (long)blockIdx.x * blockDim.x + threadIdx.x;
  int e = (int)(tid >> 4), hh = (int)(tid & 15);
  if (e >= Etot) return;
  int s, d;
  if (e < E) { s = ei[e]; d = ei[E + e]; } else { s = d = e - E; }
  float mx = ord2f(lmax[(size_t)d * 16 + hh]);
  float ex = __expf(logit[(size_t)e * 16 + hh] - mx);
  atomicAdd(&den[(size_t)d * 16 + hh], ex);
  const int HC = 16 * C;
#pragma unroll
  for (int c = 0; c < C; ++c) {
    int f = hh * C + c;
    atomicAdd(&out_num[(size_t)d * HC + f], xl[(size_t)s * HC + f] * ex);
  }
}

// ---- layer-0 finish: divide + bias, f32 (pre-BatchNorm) --------------------
__global__ void k_finish64_f32(const float* __restrict__ out_num,
                               const float* __restrict__ den,
                               const float* __restrict__ bias,
                               float* __restrict__ h, int N) {
  int tid = blockIdx.x * blockDim.x + threadIdx.x;
  if (tid >= N * 64) return;
  int n = tid >> 6, f = tid & 63, hh = f >> 2;
  h[tid] = out_num[tid] / den[(size_t)n * 16 + hh] + bias[f];
}

// ---- main-layer finish: divide + bias + LeakyReLU(0.01) -> f16 activations -
__global__ void k_finish64_f16(const float* __restrict__ out_num,
                               const float* __restrict__ den,
                               const float* __restrict__ bias,
                               _Float16* __restrict__ h16, int N) {
  int tid = blockIdx.x * blockDim.x + threadIdx.x;
  if (tid >= N * 64) return;
  int n = tid >> 6, f = tid & 63, hh = f >> 2;
  float v = out_num[tid] / den[(size_t)n * 16 + hh] + bias[f];
  v = (v > 0.0f) ? v : 0.01f * v;
  h16[tid] = (_Float16)v;
}

// ---- batch-norm over nodes: one block per channel --------------------------
__global__ void k_bn_stats(const float* __restrict__ h,
                           float* __restrict__ sums, float* __restrict__ sumsq, int N) {
  int c = blockIdx.x;
  float s = 0.0f, sq = 0.0f;
  for (int n = threadIdx.x; n < N; n += blockDim.x) {
    float v = h[(size_t)n * 64 + c];
    s += v; sq += v * v;
  }
  __shared__ float sh[256], shq[256];
  sh[threadIdx.x] = s; shq[threadIdx.x] = sq; __syncthreads();
  for (int st = 128; st > 0; st >>= 1) {
    if (threadIdx.x < st) { sh[threadIdx.x] += sh[threadIdx.x + st];
                            shq[threadIdx.x] += shq[threadIdx.x + st]; }
    __syncthreads();
  }
  if (threadIdx.x == 0) { sums[c] = sh[0]; sumsq[c] = shq[0]; }
}

__global__ void k_bn_apply(const float* __restrict__ h,
                           const float* __restrict__ sums, const float* __restrict__ sumsq,
                           const float* __restrict__ gamma, const float* __restrict__ beta,
                           _Float16* __restrict__ h16, int N) {
  int tid = blockIdx.x * blockDim.x + threadIdx.x;
  if (tid >= N * 64) return;
  int c = tid & 63;
  float invN = 1.0f / (float)N;
  float mu  = sums[c] * invN;
  float var = sumsq[c] * invN - mu * mu;
  float v = (h[tid] - mu) * rsqrtf(var + 1e-5f) * gamma[c] + beta[c];
  v = (v > 0.0f) ? v : 0.01f * v;               // LeakyReLU(0.01)
  h16[tid] = (_Float16)v;
}

// ---- head-mean for concat=False layers -------------------------------------
__global__ void k_meanheads(const float* __restrict__ out_num,
                            const float* __restrict__ den,
                            const float* __restrict__ bias,
                            float* __restrict__ out, int N) {
  int n = blockIdx.x * blockDim.x + threadIdx.x;
  if (n >= N) return;
  float s = 0.0f;
#pragma unroll
  for (int hh = 0; hh < 16; ++hh)
    s += out_num[(size_t)n * 16 + hh] / den[(size_t)n * 16 + hh];
  out[n] = s * (1.0f / 16.0f) + bias[0];
}

// ---- masked per-graph mean --------------------------------------------------
__global__ void k_sat_accum(const float* __restrict__ fin,
                            const int* __restrict__ mask, const int* __restrict__ batch,
                            float* __restrict__ gsum, float* __restrict__ gcnt, int N) {
  int n = blockIdx.x * blockDim.x + threadIdx.x;
  if (n >= N) return;
  if (mask[n] == 0) {
    int g = batch[n];
    atomicAdd(&gsum[g], fin[n]);
    atomicAdd(&gcnt[g], 1.0f);
  }
}

__global__ void k_sat_finish(const float* __restrict__ gsum,
                             const float* __restrict__ gcnt,
                             float* __restrict__ out, int G) {
  int g = threadIdx.x;
  if (g < G) out[g] = gsum[g] / fmaxf(gcnt[g], 1.0f);
}

// ---------------------------------------------------------------------------
extern "C" void kernel_launch(void* const* d_in, const int* in_sizes, int n_in,
                              void* d_out, int out_size, void* d_ws, size_t ws_size,
                              hipStream_t stream) {
  (void)in_sizes; (void)n_in; (void)out_size; (void)ws_size;
  const int N = NNODES, E = NEDGES, G = NGRAPH;
  const int Etot = E + N;

  const float* x         = (const float*)d_in[0];
  const float* edge_attr = (const float*)d_in[1];
  const int*   ei        = (const int*)d_in[2];
  const int*   mask      = (const int*)d_in[3];
  const int*   batch     = (const int*)d_in[4];
  const float* g0_Wl = (const float*)d_in[5];
  const float* g0_bl = (const float*)d_in[6];
  const float* g0_Wr = (const float*)d_in[7];
  const float* g0_br = (const float*)d_in[8];
  const float* g0_We = (const float*)d_in[9];
  const float* g0_att = (const float*)d_in[10];
  const float* g0_bias = (const float*)d_in[11];
  const float* bn_gamma = (const float*)d_in[12];
  const float* bn_beta  = (const float*)d_in[13];
  const float* m_Wl = (const float*)d_in[14];
  const float* m_bl = (const float*)d_in[15];
  const float* m_Wr = (const float*)d_in[16];
  const float* m_br = (const float*)d_in[17];
  const float* m_We = (const float*)d_in[18];
  const float* m_att = (const float*)d_in[19];
  const float* m_bias = (const float*)d_in[20];
  const float* f_Wl = (const float*)d_in[21];
  const float* f_bl = (const float*)d_in[22];
  const float* f_Wr = (const float*)d_in[23];
  const float* f_br = (const float*)d_in[24];
  const float* f_We = (const float*)d_in[25];
  const float* f_att = (const float*)d_in[26];
  const float* f_bias = (const float*)d_in[27];
  const float* s_Wl = (const float*)d_in[28];
  const float* s_bl = (const float*)d_in[29];
  const float* s_Wr = (const float*)d_in[30];
  const float* s_br = (const float*)d_in[31];
  const float* s_We = (const float*)d_in[32];
  const float* s_att = (const float*)d_in[33];
  const float* s_bias = (const float*)d_in[34];

  // ---- workspace carve (floats), 64-element (256B) aligned ----
  float* base = (float*)d_ws;
  size_t off = 0;
  auto take = [&](size_t nel) {
    float* p = base + off;
    off += (nel + 63) & ~(size_t)63;
    return p;
  };
  float*     cnt       = take(N);
  float*     esum      = take((size_t)2 * N);
  float*     loop_attr = take((size_t)2 * N);
  float*     hA        = take((size_t)64 * N);          // layer-0 pre-BN f32
  _Float16*  h16       = (_Float16*)take((size_t)32 * NPAD); // [NPAD,64] f16
  float*     xl        = take((size_t)64 * NPAD);       // padded rows
  float*     xr        = take((size_t)64 * NPAD);
  float*     out_num   = take((size_t)64 * N);
  float*     den       = take((size_t)16 * N);
  unsigned*  lmax      = (unsigned*)take((size_t)16 * N);
  float*     logit     = take((size_t)16 * Etot);
  _Float16*  wlt       = (_Float16*)take(2048);         // 64x64 f16 transposed
  _Float16*  wrt       = (_Float16*)take(2048);
  float*     bn_sums   = take(64);
  float*     bn_sumsq  = take(64);
  float*     gsum      = take(64);
  float*     gcnt      = take(64);
  float*     fin       = take(N);

  const int TB = 256;
  const int nf64_blocks = (N * 64 + TB - 1) / TB;
  const int node_blocks = (N + TB - 1) / TB;
  const long eth = (long)Etot * 16;
  const int eblocks = (int)((eth + TB - 1) / TB);
  const int g64_blocks = (N + 15) / 16;   // 16-row tiles, max row 25008 <= NPAD
  const int g16_blocks = (N + 63) / 64;   // 64-row tiles, max row 25024 <= NPAD

  auto zero_edge_state = [&](int C) {
    hipMemsetAsync((void*)lmax, 0, (size_t)16 * N * sizeof(unsigned), stream);
    hipMemsetAsync((void*)den, 0, (size_t)16 * N * sizeof(float), stream);
    hipMemsetAsync((void*)out_num, 0, (size_t)16 * C * N * sizeof(float), stream);
  };

  // ---- self-loop edge attributes (fill_value='mean') ----
  hipMemsetAsync((void*)cnt, 0, (size_t)N * sizeof(float), stream);
  hipMemsetAsync((void*)esum, 0, (size_t)2 * N * sizeof(float), stream);
  k_selfloop_accum<<<(E + TB - 1) / TB, TB, 0, stream>>>(ei, edge_attr, cnt, esum, E);
  k_selfloop_finish<<<node_blocks, TB, 0, stream>>>(cnt, esum, loop_attr, N);

  // ---- layer 0: GATv2(2 -> 64) + BatchNorm + LeakyReLU(0.01) ----
  k_transform2<<<nf64_blocks, TB, 0, stream>>>(x, g0_Wl, g0_bl, g0_Wr, g0_br, xl, xr, N);
  zero_edge_state(4);
  k_edge_logit<4><<<eblocks, TB, 0, stream>>>(ei, edge_attr, loop_attr, xl, xr,
                                              g0_We, g0_att, logit, lmax, E, Etot);
  k_edge_soft<4><<<eblocks, TB, 0, stream>>>(ei, logit, lmax, xl, den, out_num, E, Etot);
  k_finish64_f32<<<nf64_blocks, TB, 0, stream>>>(out_num, den, g0_bias, hA, N);
  hipMemsetAsync((void*)bn_sums, 0, 64 * sizeof(float), stream);
  hipMemsetAsync((void*)bn_sumsq, 0, 64 * sizeof(float), stream);
  k_bn_stats<<<64, TB, 0, stream>>>(hA, bn_sums, bn_sumsq, N);
  k_bn_apply<<<nf64_blocks, TB, 0, stream>>>(hA, bn_sums, bn_sumsq,
                                             bn_gamma, bn_beta, h16, N);

  // ---- 10 main GATv2(64 -> 64) layers (WMMA node transforms) ----
  for (int i = 0; i < NITER; ++i) {
    k_prep_w64<<<16, TB, 0, stream>>>(m_Wl + (size_t)i * 4096,
                                      m_Wr + (size_t)i * 4096, wlt, wrt);
    k_gemm_node64<<<g64_blocks, TB, 0, stream>>>(h16, wlt, m_bl + (size_t)i * 64,
                                                 wrt, m_br + (size_t)i * 64, xl, xr);
    zero_edge_state(4);
    k_edge_logit<4><<<eblocks, TB, 0, stream>>>(ei, edge_attr, loop_attr, xl, xr,
                                                m_We + (size_t)i * 128,
                                                m_att + (size_t)i * 64,
                                                logit, lmax, E, Etot);
    k_edge_soft<4><<<eblocks, TB, 0, stream>>>(ei, logit, lmax, xl, den, out_num, E, Etot);
    k_finish64_f16<<<nf64_blocks, TB, 0, stream>>>(out_num, den,
                                                   m_bias + (size_t)i * 64, h16, N);
  }

  float* out_f = (float*)d_out;

  // ---- sbs head: GATv2(64 -> 1, heads=16, concat=False) -> mus = d_out[0:N) ----
  k_prep_w16<<<4, TB, 0, stream>>>(s_Wl, s_Wr, wlt, wrt);
  k_gemm_node16<<<g16_blocks, TB, 0, stream>>>(h16, wlt, s_bl, wrt, s_br, xl, xr);
  zero_edge_state(1);
  k_edge_logit<1><<<eblocks, TB, 0, stream>>>(ei, edge_attr, loop_attr, xl, xr,
                                              s_We, s_att, logit, lmax, E, Etot);
  k_edge_soft<1><<<eblocks, TB, 0, stream>>>(ei, logit, lmax, xl, den, out_num, E, Etot);
  k_meanheads<<<node_blocks, TB, 0, stream>>>(out_num, den, s_bias, out_f, N);

  // ---- final head: GATv2(64 -> 1) -> masked per-graph mean -> d_out[N:N+G) ----
  k_prep_w16<<<4, TB, 0, stream>>>(f_Wl, f_Wr, wlt, wrt);
  k_gemm_node16<<<g16_blocks, TB, 0, stream>>>(h16, wlt, f_bl, wrt, f_br, xl, xr);
  zero_edge_state(1);
  k_edge_logit<1><<<eblocks, TB, 0, stream>>>(ei, edge_attr, loop_attr, xl, xr,
                                              f_We, f_att, logit, lmax, E, Etot);
  k_edge_soft<1><<<eblocks, TB, 0, stream>>>(ei, logit, lmax, xl, den, out_num, E, Etot);
  k_meanheads<<<node_blocks, TB, 0, stream>>>(out_num, den, f_bias, fin, N);

  hipMemsetAsync((void*)gsum, 0, 64 * sizeof(float), stream);
  hipMemsetAsync((void*)gcnt, 0, 64 * sizeof(float), stream);
  k_sat_accum<<<node_blocks, TB, 0, stream>>>(fin, mask, batch, gsum, gcnt, N);
  k_sat_finish<<<1, 64, 0, stream>>>(gsum, gcnt, out_f + N, G);
}
